// BitLinear_52905407152476
// MI455X (gfx1250) — 8-bit matrix operand order repaired and checked by reading, compiled, not run on hardware
//
#include <hip/hip_runtime.h>

typedef __attribute__((ext_vector_type(8))) int v8i;

#define EPSQ   1e-5f
#define DDIM   1024
#define MTOT   32768
#define LSTR   80

__global__ void k_wabs(const float* __restrict__ w, float* __restrict__ partial) {
    __shared__ float sd[256];
    const int t = threadIdx.x;
    const size_t i = ((size_t)blockIdx.x * 256 + t) * 4;
    float4 v = *(const float4*)(w + i);
    sd[t] = fabsf(v.x) + fabsf(v.y) + fabsf(v.z) + fabsf(v.w);
    __syncthreads();
    for (int o = 128; o > 0; o >>= 1) {
        if (t < o) sd[t] += sd[t + o];
        __syncthreads();
    }
    if (t == 0) partial[blockIdx.x] = sd[0];
}

__global__ void k_wscale(const float* __restrict__ partial, float* __restrict__ scal) {
    __shared__ float sd[256];
    const int t = threadIdx.x;
    sd[t] = partial[t] + partial[t + 256] + partial[t + 512] + partial[t + 768];
    __syncthreads();
    for (int o = 128; o > 0; o >>= 1) {
        if (t < o) sd[t] += sd[t + o];
        __syncthreads();
    }
    if (t == 0) {
        float s = fmaxf(sd[0] * (1.0f / 1048576.0f), EPSQ);
        scal[0] = s;
        scal[1] = 1.0f / s;
    }
}

__global__ void k_wq(const float* __restrict__ w, char* __restrict__ qw,
                     const float* __restrict__ scal) {
    const float inv_s = scal[1];
    const size_t i = ((size_t)blockIdx.x * 256 + threadIdx.x) * 4;
    float4 v = *(const float4*)(w + i);
    int q0 = (int)rintf(v.x * inv_s); q0 = q0 < -1 ? -1 : (q0 > 1 ? 1 : q0);
    int q1 = (int)rintf(v.y * inv_s); q1 = q1 < -1 ? -1 : (q1 > 1 ? 1 : q1);
    int q2 = (int)rintf(v.z * inv_s); q2 = q2 < -1 ? -1 : (q2 > 1 ? 1 : q2);
    int q3 = (int)rintf(v.w * inv_s); q3 = q3 < -1 ? -1 : (q3 > 1 ? 1 : q3);
    unsigned p = (unsigned)(q0 & 255) | ((unsigned)(q1 & 255) << 8) |
                 ((unsigned)(q2 & 255) << 16) | ((unsigned)(q3 & 255) << 24);
    ((unsigned*)qw)[i >> 2] = p;
}

__global__ void k_aq(const float* __restrict__ x, char* __restrict__ qx,
                     float* __restrict__ rowf, const float* __restrict__ scal) {
    const int token = blockIdx.x * 8 + (threadIdx.x >> 5);
    const int lane  = threadIdx.x & 31;
    const float* row = x + (size_t)token * DDIM;

    float4 v[8];
    float mx = 0.0f;
#pragma unroll
    for (int c = 0; c < 8; ++c) {
        v[c] = *(const float4*)(row + lane * 4 + c * 128);
        mx = fmaxf(mx, fmaxf(fmaxf(fabsf(v[c].x), fabsf(v[c].y)),
                             fmaxf(fabsf(v[c].z), fabsf(v[c].w))));
    }
#pragma unroll
    for (int off = 16; off > 0; off >>= 1)
        mx = fmaxf(mx, __shfl_xor(mx, off, 32));

    const float maxc  = fmaxf(mx, EPSQ);
    const float scale = 127.0f / maxc;
    unsigned* qrow = (unsigned*)(qx + (size_t)token * DDIM);
#pragma unroll
    for (int c = 0; c < 8; ++c) {
        int q0 = (int)rintf(v[c].x * scale); q0 = q0 < -128 ? -128 : (q0 > 127 ? 127 : q0);
        int q1 = (int)rintf(v[c].y * scale); q1 = q1 < -128 ? -128 : (q1 > 127 ? 127 : q1);
        int q2 = (int)rintf(v[c].z * scale); q2 = q2 < -128 ? -128 : (q2 > 127 ? 127 : q2);
        int q3 = (int)rintf(v[c].w * scale); q3 = q3 < -128 ? -128 : (q3 > 127 ? 127 : q3);
        qrow[lane + c * 32] = (unsigned)(q0 & 255) | ((unsigned)(q1 & 255) << 8) |
                              ((unsigned)(q2 & 255) << 16) | ((unsigned)(q3 & 255) << 24);
    }
    if (lane == 0) rowf[token] = maxc * scal[0] * (1.0f / 127.0f);
}

__global__ __launch_bounds__(256) void k_gemm(const char* __restrict__ qx,
                                              const char* __restrict__ qw,
                                              const float* __restrict__ rowf,
                                              float* __restrict__ out) {
    __shared__ char lds[2][2][128 * LSTR];

    const int tid    = threadIdx.x;
    const int lane   = tid & 31;
    const int wid    = tid >> 5;
    const int wave_m = wid >> 2;
    const int wave_n = wid & 3;
    const int half   = lane >> 4;
    const int l15    = lane & 15;
    const int bm     = blockIdx.y * 128;
    const int bn     = blockIdx.x * 128;

    const int c0 = tid, c1 = tid + 256;
    const int r0 = c0 >> 2, o0 = (c0 & 3) * 16;
    const int r1 = c1 >> 2, o1 = (c1 & 3) * 16;
    const char* aptr0 = qx + (size_t)(bm + r0) * DDIM + o0;
    const char* aptr1 = qx + (size_t)(bm + r1) * DDIM + o1;
    const char* bptr0 = qw + (size_t)(bn + r0) * DDIM + o0;
    const char* bptr1 = qw + (size_t)(bn + r1) * DDIM + o1;

    v8i acc[4][2] = {};

    int4 ar0 = *(const int4*)(aptr0);
    int4 ar1 = *(const int4*)(aptr1);
    int4 br0 = *(const int4*)(bptr0);
    int4 br1 = *(const int4*)(bptr1);

    for (int kt = 0; kt < DDIM / 64; ++kt) {
        char* As = &lds[kt & 1][0][0];
        char* Bs = &lds[kt & 1][1][0];
        *(int4*)(As + r0 * LSTR + o0) = ar0;
        *(int4*)(As + r1 * LSTR + o1) = ar1;
        *(int4*)(Bs + r0 * LSTR + o0) = br0;
        *(int4*)(Bs + r1 * LSTR + o1) = br1;
        __syncthreads();

        if (kt + 1 < DDIM / 64) {
            const int ko = (kt + 1) * 64;
            ar0 = *(const int4*)(aptr0 + ko);
            ar1 = *(const int4*)(aptr1 + ko);
            br0 = *(const int4*)(bptr0 + ko);
            br1 = *(const int4*)(bptr1 + ko);
        }
        if (kt + 3 < DDIM / 64) {
            const int kp = (kt + 3) * 64;
            __builtin_prefetch(aptr0 + kp, 0, 0);
            __builtin_prefetch(bptr0 + kp, 0, 0);
        }

        v8i afr[4];
#pragma unroll
        for (int mt = 0; mt < 4; ++mt) {
            const char* rp = As + (wave_m * 64 + mt * 16 + l15) * LSTR + 16 * half;
            uint2 q0 = *(const uint2*)(rp);
            uint2 q1 = *(const uint2*)(rp + 8);
            uint2 q2 = *(const uint2*)(rp + 32);
            uint2 q3 = *(const uint2*)(rp + 40);
            v8i a;
            a[0] = (int)q0.x; a[1] = (int)q0.y; a[2] = (int)q1.x; a[3] = (int)q1.y;
            a[4] = (int)q2.x; a[5] = (int)q2.y; a[6] = (int)q3.x; a[7] = (int)q3.y;
            afr[mt] = a;
        }
        v8i bfr[2];
#pragma unroll
        for (int nt = 0; nt < 2; ++nt) {
            const char* rp = Bs + (wave_n * 32 + nt * 16 + l15) * LSTR + 16 * half;
            int4 p0 = *(const int4*)(rp);
            int4 p1 = *(const int4*)(rp + 32);
            v8i b;
            b[0] = p0.x; b[1] = p0.y; b[2] = p0.z; b[3] = p0.w;
            b[4] = p1.x; b[5] = p1.y; b[6] = p1.z; b[7] = p1.w;
            bfr[nt] = b;
        }

#pragma unroll
        for (int mt = 0; mt < 4; ++mt)
#pragma unroll
            for (int nt = 0; nt < 2; ++nt)
                acc[mt][nt] = __builtin_amdgcn_wmma_i32_16x16x64_iu8(
                     true, afr[mt],  true, bfr[nt],
                    acc[mt][nt],  false,  false);
    }

#pragma unroll
    for (int mt = 0; mt < 4; ++mt) {
        const int mbase = bm + wave_m * 64 + mt * 16 + 8 * half;
        float rf[8];
#pragma unroll
        for (int r = 0; r < 8; ++r) rf[r] = rowf[mbase + r];
#pragma unroll
        for (int nt = 0; nt < 2; ++nt) {
            const int n = bn + wave_n * 32 + nt * 16 + l15;
#pragma unroll
            for (int r = 0; r < 8; ++r)
                out[(size_t)(mbase + r) * DDIM + n] = (float)acc[mt][nt][r] * rf[r];
        }
    }
}

extern "C" void kernel_launch(void* const* d_in, const int* in_sizes, int n_in,
                              void* d_out, int out_size, void* d_ws, size_t ws_size,
                              hipStream_t stream) {
    const float* x = (const float*)d_in[0];
    const float* w = (const float*)d_in[1];
    float* out = (float*)d_out;
    char* ws = (char*)d_ws;

    char*  qx      = ws;
    char*  qw      = ws + (size_t)MTOT * DDIM;
    float* rowf    = (float*)(qw + (size_t)DDIM * DDIM);
    float* partial = rowf + MTOT;
    float* scal    = partial + 1024;

    k_wabs  <<<1024, 256, 0, stream>>>(w, partial);
    k_wscale<<<   1, 256, 0, stream>>>(partial, scal);
    k_wq    <<<1024, 256, 0, stream>>>(w, qw, scal);
    k_aq    <<<4096, 256, 0, stream>>>(x, qx, rowf, scal);
    k_gemm  <<<dim3(8, 256), 256, 0, stream>>>(qx, qw, rowf, out);
}
